// SimpleCrossAttention_75161927680538
// MI455X (gfx1250) — matrix operand order repaired and checked by reading, not compiled, not run on hardware
//
#include <hip/hip_runtime.h>
#include <hip/hip_bf16.h>
#include <math.h>

#define DIM     256
#define NHEADS  4
#define HDIM    64
#define NQ      4096
#define NK      4096
#define NGRAPHS 64

typedef __attribute__((ext_vector_type(16))) __bf16 v16bf;
typedef __attribute__((ext_vector_type(8)))  __bf16 v8bf;
typedef __attribute__((ext_vector_type(8)))  float  v8f;
typedef unsigned short bfu;

static __device__ __forceinline__ __bf16 bfu2b(bfu u) {
  return __builtin_bit_cast(__bf16, u);
}
static __device__ __forceinline__ bfu f2bfu(float f) {
  union { float f; unsigned u; } v; v.f = f;
  unsigned r = v.u + 0x7FFFu + ((v.u >> 16) & 1u);   // round-to-nearest-even
  return (bfu)(r >> 16);
}
static __device__ __forceinline__ __bf16 f2b(float f) { return bfu2b(f2bfu(f)); }
static __device__ __forceinline__ unsigned pack2(float lo, float hi) {
  return (unsigned)f2bfu(lo) | ((unsigned)f2bfu(hi) << 16);
}

static __device__ __forceinline__ v16bf concat16(v8bf a, v8bf b) {
  return __builtin_shufflevector(a, b, 0, 1, 2, 3, 4, 5, 6, 7,
                                       8, 9, 10, 11, 12, 13, 14, 15);
}

// A fragment (16x32 bf16): per lane one contiguous 32B chunk at K=hi*16.
static __device__ __forceinline__ v16bf load_a(const bfu* p, int hi) {
  return *(const v16bf*)(p + hi * 16);
}
// B fragment (32x16 bf16): per lane one contiguous 32B chunk at K=hi*16.
static __device__ __forceinline__ v16bf load_b(const bfu* p, int hi) {
  return *(const v16bf*)(p + hi * 16);
}

static __device__ __forceinline__ v8f wmma_bf16(v16bf a, v16bf b, v8f c) {
  return __builtin_amdgcn_wmma_f32_16x16x32_bf16(false, a, false, b,
                                                 (short)0, c, false, false);
}

// ---------------------------------------------------------------- ranges ----
static __device__ __forceinline__ int lowerb(const int* a, int n, int v) {
  int lo = 0, hi = n;
  while (lo < hi) { int mid = (lo + hi) >> 1; if (a[mid] < v) lo = mid + 1; else hi = mid; }
  return lo;
}

__global__ void ranges_kernel(const int* __restrict__ qb, const int* __restrict__ kb,
                              int* __restrict__ r) {
  int g = threadIdx.x;
  if (g < NGRAPHS) {
    r[g]               = lowerb(qb, NQ, g);
    r[NGRAPHS + g]     = lowerb(qb, NQ, g + 1);
    r[2 * NGRAPHS + g] = lowerb(kb, NK, g);
    r[3 * NGRAPHS + g] = lowerb(kb, NK, g + 1);
  }
}

// ------------------------------------------------------------------ zero ----
__global__ void zero_kernel(float4* __restrict__ p, int n4) {
  int i = blockIdx.x * blockDim.x + threadIdx.x;
  if (i < n4) p[i] = make_float4(0.f, 0.f, 0.f, 0.f);
}

// ----------------------------------------------------- fp32 -> bf16 (pairs) ----
__global__ __launch_bounds__(256) void cvt_kernel(const float* __restrict__ s,
                                                  unsigned* __restrict__ d, int n2) {
  int i = blockIdx.x * blockDim.x + threadIdx.x;
  if (i < n2) {
    float2 v = ((const float2*)s)[i];
    d[i] = pack2(v.x, v.y);
  }
}

// ------------------------------------------------------------ projections ----
// grid: (32 row-blocks of 128 rows, head, matrix{Q,K,V}); block: 256 = 8 waves
__global__ __launch_bounds__(256) void proj_kernel(
    const bfu* __restrict__ qfB, const bfu* __restrict__ kfB,
    const bfu* __restrict__ WqB, const bfu* __restrict__ WkB, const bfu* __restrict__ WvB,
    const float* __restrict__ bq, const float* __restrict__ bk, const float* __restrict__ bv,
    bfu* __restrict__ Qb, bfu* __restrict__ Kb, bfu* __restrict__ Vt) {
  int wave = threadIdx.x >> 5, lane = threadIdx.x & 31;
  int n = lane & 15, hi = lane >> 4;
  int h = blockIdx.y, mat = blockIdx.z;
  int row0 = blockIdx.x * 128 + wave * 16;

  const bfu* X    = (mat == 0) ? qfB : kfB;
  const bfu* W    = (mat == 0 ? WqB : (mat == 1 ? WkB : WvB)) + (size_t)h * HDIM * DIM;
  const float* bias = (mat == 0 ? bq : (mat == 1 ? bk : bv)) + (size_t)h * HDIM;

  v8f acc[4] = {{}, {}, {}, {}};
  const bfu* arow = X + (size_t)(row0 + n) * DIM;
  for (int kb0 = 0; kb0 < DIM; kb0 += 32) {
    v16bf A = load_a(arow + kb0, hi);
    #pragma unroll
    for (int ct = 0; ct < 4; ++ct) {
      v16bf B = load_b(W + (size_t)(ct * 16 + n) * DIM + kb0, hi);
      acc[ct] = wmma_bf16(A, B, acc[ct]);
    }
  }
  if (mat < 2) {
    bfu* Out = (mat == 0 ? Qb : Kb) + (size_t)h * NQ * HDIM;
    #pragma unroll
    for (int ct = 0; ct < 4; ++ct)
      #pragma unroll
      for (int j = 0; j < 8; ++j) {
        int e = ct * 16 + n;
        Out[(size_t)(row0 + hi * 8 + j) * HDIM + e] = f2bfu(acc[ct][j] + bias[e]);
      }
  } else {  // V stored transposed: Vt[h][e][k]
    bfu* Out = Vt + (size_t)h * HDIM * NK;
    #pragma unroll
    for (int ct = 0; ct < 4; ++ct)
      #pragma unroll
      for (int j = 0; j < 8; ++j) {
        int e = ct * 16 + n;
        Out[(size_t)e * NK + (row0 + hi * 8 + j)] = f2bfu(acc[ct][j] + bias[e]);
      }
  }
}

// S^T tile with preloaded Q fragments:
// C[krel=hi*8+j (rows), q=lane&15 (cols)] = K(16xE) . Q^T(Ex16)
static __device__ __forceinline__ v8f st_tile(const bfu* __restrict__ Kh,
                                              int kt, v16bf Bq0, v16bf Bq1,
                                              int lane) {
  int n = lane & 15, hi = lane >> 4;
  const bfu* kr = Kh + (size_t)min(kt + n, NK - 1) * HDIM;
  v8f c = {};
  c = wmma_bf16(load_a(kr, hi), Bq0, c);
  c = wmma_bf16(load_a(kr + 32, hi), Bq1, c);
  return c;
}

// ------------------------------------------------------------- attention ----
// grid: (64 graphs, 8 q-strips); block: 128 threads = 4 waves (wave == head)
__global__ __launch_bounds__(128) void attn_kernel(
    const bfu* __restrict__ Qb, const bfu* __restrict__ Kb, const bfu* __restrict__ Vt,
    const int* __restrict__ r, bfu* __restrict__ OcB, float* __restrict__ attn) {
  __shared__ float s_acc[16 * 32];
  int g = blockIdx.x;
  int qs = r[g], qe = r[NGRAPHS + g];
  int ks = r[2 * NGRAPHS + g], ke = r[3 * NGRAPHS + g];
  int h = threadIdx.x >> 5, lane = threadIdx.x & 31;
  int n = lane & 15, hi = lane >> 4;
  const bfu* Qh = Qb + (size_t)h * NQ * HDIM;
  const bfu* Kh = Kb + (size_t)h * NQ * HDIM;
  const bfu* Vh = Vt + (size_t)h * HDIM * NK;   // [e][k]
  const float scale = 0.125f;  // 1/sqrt(64)
  const int kt16 = ks & ~15, kt32 = ks & ~31;   // aligned key-tile starts

  for (int q0 = qs + 16 * blockIdx.y; q0 < qe; q0 += 16 * 8) {
    // hoisted Q fragments for this 16-query tile (loop-invariant over keys)
    const bfu* qr = Qh + (size_t)min(q0 + n, NQ - 1) * HDIM;
    v16bf Bq0 = load_b(qr, hi);
    v16bf Bq1 = load_b(qr + 32, hi);

    // -------- pass 1: online (m, l) per q column (this lane's column = n) ---
    float m = -INFINITY, l = 0.f;
    for (int kt = kt16; kt < ke; kt += 16) {
      v8f c = st_tile(Kh, kt, Bq0, Bq1, lane);
      float tmax = -INFINITY;
      #pragma unroll
      for (int j = 0; j < 8; ++j) {
        int k = kt + hi * 8 + j;
        if (k >= ks && k < ke) tmax = fmaxf(tmax, c[j] * scale);
      }
      float rmax = fmaxf(tmax, __shfl_xor(tmax, 16, 32));
      if (rmax > -INFINITY) {
        float newm = fmaxf(m, rmax);
        float ps = 0.f;
        #pragma unroll
        for (int j = 0; j < 8; ++j) {
          int k = kt + hi * 8 + j;
          if (k >= ks && k < ke) ps += __expf(c[j] * scale - newm);
        }
        float fs = ps + __shfl_xor(ps, 16, 32);
        l = l * __expf(m - newm) + fs;
        m = newm;
      }
    }
    float invl = (l > 0.f) ? 1.f / l : 0.f;

    // -------- pass 2: probabilities -> attn mean + O^T = V^T . P^T ---------
    v8f ot[4] = {{}, {}, {}, {}};
    for (int kt = kt32; kt < ke; kt += 32) {
      __builtin_prefetch(Kh + (size_t)min(kt + 32 + n, NK - 1) * HDIM, 0, 1);
      __syncthreads();
      for (int i = threadIdx.x; i < 512; i += 128) s_acc[i] = 0.f;
      __syncthreads();

      v8f c0 = st_tile(Kh, kt, Bq0, Bq1, lane);
      v8f c1 = st_tile(Kh, kt + 16, Bq0, Bq1, lane);
      float p0[8], p1[8];
      #pragma unroll
      for (int j = 0; j < 8; ++j) {
        int k0 = kt + hi * 8 + j;
        int k1 = k0 + 16;
        bool v0 = (k0 >= ks && k0 < ke), v1 = (k1 >= ks && k1 < ke);
        p0[j] = v0 ? __expf(c0[j] * scale - m) * invl : 0.f;
        p1[j] = v1 ? __expf(c1[j] * scale - m) * invl : 0.f;
        if (v0) atomicAdd(&s_acc[n * 32 + hi * 8 + j], p0[j]);
        if (v1) atomicAdd(&s_acc[n * 32 + 16 + hi * 8 + j], p1[j]);
      }
      __syncthreads();
      for (int i = threadIdx.x; i < 512; i += 128) {
        int qq = q0 + (i >> 5), kk = kt + (i & 31);
        if (qq < qe && kk >= ks && kk < ke)
          attn[(size_t)qq * NK + kk] = s_acc[i] * 0.25f;  // mean over 4 heads
      }

      // Transpose P (C-layout) into B-layout of P^T (32k x 16q)
      float full[16];
      #pragma unroll
      for (int j = 0; j < 8; ++j) {
        float o0 = __shfl_xor(p0[j], 16, 32);
        float o1 = __shfl_xor(p1[j], 16, 32);
        full[j]     = hi ? o1    : p0[j];
        full[8 + j] = hi ? p1[j] : o0;
      }
      v16bf Bp;
      #pragma unroll
      for (int t = 0; t < 16; ++t) Bp[t] = f2b(full[t]);

      // A fragments: V^T e-tiles (rows = e, K = keys) -- contiguous in Vt
      #pragma unroll
      for (int et = 0; et < 4; ++et) {
        int e = et * 16 + n;
        v16bf Av = load_a(Vh + (size_t)e * NK + kt, hi);
        ot[et] = wmma_bf16(Av, Bp, ot[et]);
      }
    }

    // fused epilogue: O^T C-layout -> bf16 Ocat[row][h*64+e], one b128 per e-tile
    int q = q0 + n;
    if (q < qe) {
      bfu* orow = OcB + (size_t)q * DIM + h * HDIM + hi * 8;
      #pragma unroll
      for (int et = 0; et < 4; ++et) {
        uint4 pk;
        pk.x = pack2(ot[et][0], ot[et][1]);
        pk.y = pack2(ot[et][2], ot[et][3]);
        pk.z = pack2(ot[et][4], ot[et][5]);
        pk.w = pack2(ot[et][6], ot[et][7]);
        *(uint4*)(orow + et * 16) = pk;
      }
    }
  }
}

// ------------------------------------------------------- output projection ----
// grid: (32 row-blocks, 4 col-groups of 64); block: 256 = 8 waves
__global__ __launch_bounds__(256) void oproj_kernel(
    const bfu* __restrict__ OcB, const bfu* __restrict__ WoB,
    const float* __restrict__ bo, float* __restrict__ out) {
  int wave = threadIdx.x >> 5, lane = threadIdx.x & 31;
  int n = lane & 15, hi = lane >> 4;
  int row0 = blockIdx.x * 128 + wave * 16;
  int cg = blockIdx.y * 64;

  v8f acc[4] = {{}, {}, {}, {}};
  const bfu* arow = OcB + (size_t)(row0 + n) * DIM;
  for (int kb0 = 0; kb0 < DIM; kb0 += 32) {
    v16bf A = load_a(arow + kb0, hi);
    #pragma unroll
    for (int ct = 0; ct < 4; ++ct) {
      v16bf B = load_b(WoB + (size_t)(cg + ct * 16 + n) * DIM + kb0, hi);
      acc[ct] = wmma_bf16(A, B, acc[ct]);
    }
  }
  #pragma unroll
  for (int ct = 0; ct < 4; ++ct)
    #pragma unroll
    for (int j = 0; j < 8; ++j) {
      int o = cg + ct * 16 + n;
      out[(size_t)(row0 + hi * 8 + j) * DIM + o] = acc[ct][j] + bo[o];
    }
}

// ------------------------------------------------------------------ launch ----
extern "C" void kernel_launch(void* const* d_in, const int* in_sizes, int n_in,
                              void* d_out, int out_size, void* d_ws, size_t ws_size,
                              hipStream_t stream) {
  (void)in_sizes; (void)n_in; (void)out_size; (void)ws_size;

  const float* q_feat  = (const float*)d_in[0];
  const float* k_feat  = (const float*)d_in[1];
  const int*   q_batch = (const int*)d_in[2];
  const int*   k_batch = (const int*)d_in[3];
  const float* Wq = (const float*)d_in[4];
  const float* bq = (const float*)d_in[5];
  const float* Wk = (const float*)d_in[6];
  const float* bk = (const float*)d_in[7];
  const float* Wv = (const float*)d_in[8];
  const float* bv = (const float*)d_in[9];
  const float* Wo = (const float*)d_in[10];
  const float* bo = (const float*)d_in[11];

  float* out  = (float*)d_out;              // [4096 x 256]
  float* attn = out + (size_t)NQ * DIM;     // [4096 x 4096] attn mean

  // workspace layout (~12.5 MB)
  char* ws = (char*)d_ws;
  bfu* qfB    = (bfu*)(ws + 0x000000);   // 2 MB
  bfu* kfB    = (bfu*)(ws + 0x200000);   // 2 MB
  bfu* WqB    = (bfu*)(ws + 0x400000);   // 128 KB
  bfu* WkB    = (bfu*)(ws + 0x420000);
  bfu* WvB    = (bfu*)(ws + 0x440000);
  bfu* WoB    = (bfu*)(ws + 0x460000);
  bfu* Qb     = (bfu*)(ws + 0x480000);   // 2 MB
  bfu* Kb     = (bfu*)(ws + 0x680000);   // 2 MB
  bfu* Vt     = (bfu*)(ws + 0x880000);   // 2 MB (transposed V)
  bfu* OcB    = (bfu*)(ws + 0xA80000);   // 2 MB (concat head outputs, bf16)
  int* ranges = (int*)(ws + 0xC80000);   // 1 KB

  auto cvt = [&](const float* s, bfu* d, int nelem) {
    int n2 = nelem / 2;
    cvt_kernel<<<(n2 + 255) / 256, 256, 0, stream>>>(s, (unsigned*)d, n2);
  };

  ranges_kernel<<<1, 64, 0, stream>>>(q_batch, k_batch, ranges);
  int n4 = NQ * NK / 4;
  zero_kernel<<<(n4 + 255) / 256, 256, 0, stream>>>((float4*)attn, n4);

  cvt(q_feat, qfB, NQ * DIM);
  cvt(k_feat, kfB, NK * DIM);
  cvt(Wq, WqB, NHEADS * HDIM * DIM);
  cvt(Wk, WkB, NHEADS * HDIM * DIM);
  cvt(Wv, WvB, NHEADS * HDIM * DIM);
  cvt(Wo, WoB, DIM * DIM);

  proj_kernel<<<dim3(32, NHEADS, 3), 256, 0, stream>>>(
      qfB, kfB, WqB, WkB, WvB, bq, bk, bv, Qb, Kb, Vt);

  attn_kernel<<<dim3(NGRAPHS, 8), 128, 0, stream>>>(Qb, Kb, Vt, ranges, OcB, attn);

  oproj_kernel<<<dim3(32, 4), 256, 0, stream>>>(OcB, WoB, bo, out);
}
